// Seq2SeqWithAttention_90340342104345
// MI455X (gfx1250) — matrix operand order repaired and checked by reading, not compiled, not run on hardware
//
#include <hip/hip_runtime.h>
#include <hip/hip_bf16.h>
#include <math.h>

// ---------------- problem constants ----------------
#define BN    64      // batch
#define SSRC  128     // source length
#define TDEC  63      // decoder steps (concat(SOS, trg[:,1:63]))
#define EMB   512
#define HID   1024
#define G4H   4096    // 4*H gates

typedef __bf16 bf_t;
typedef bf_t  v16bf __attribute__((ext_vector_type(16)));
typedef float v8f   __attribute__((ext_vector_type(8)));

__device__ __forceinline__ bf_t f2bf(float f) {
    union { float f; unsigned u; } in; in.f = f;
    unsigned u = in.u;
    u += 0x7FFFu + ((u >> 16) & 1u);          // round-to-nearest-even
    union { unsigned short s; bf_t b; } out;
    out.s = (unsigned short)(u >> 16);
    return out.b;
}

__device__ __forceinline__ float sigm(float x) { return 1.f / (1.f + __expf(-x)); }
// tanh(x) = 2*sigmoid(2x) - 1  (stays on the TRANS pipe via v_exp_f32)
__device__ __forceinline__ float tanh_fast(float x) {
    return 2.f / (1.f + __expf(-2.f * x)) - 1.f;
}

// K-swizzle: swap bits 3<->4 (self-inverse).  Storing A-side data with the K
// index swizzled makes each lane's 16-element WMMA A-fragment one contiguous
// aligned 32-byte read at (row + k0 + half*16):
//   storage [k0..k0+15]  = original K [k0+0..7, k0+16..23]   (lanes 0-15)
//   storage [k0+16..+31] = original K [k0+8..15, k0+24..31]  (lanes 16-31)
__device__ __forceinline__ int swz34(int i) {
    return (i & ~24) | ((i & 8) << 1) | ((i & 16) >> 1);
}

// ---------------- fused LSTM step: gate GEMM + cell ----------------
// Wave = 1 m-tile (16 batch rows) x j-strip of 32 replicated over the 4 gate
// blocks (8 N-fragments at n = g*1024 + jb + nt*16 + nl).  All four gate
// values for a given (b, j) land in the same lane/VGPR slot -> LSTM cell is
// in-register VALU after the K loop.  K loop is split into an x-phase and an
// h-phase with compile-time bounds (IN_DIM template) so all inner-loop
// addresses are pointer + immediate offset.
// grid = 64 blocks x 64 threads (2 waves): block -> (m-half, j-strip).
template <int IN_DIM>
__global__ __launch_bounds__(64) void lstm_step_fused(
    const bf_t* __restrict__ Ax, long strideAx,  // K-swizzled x input rows
    const bf_t* __restrict__ Ah,                 // K-swizzled h in [64][1024]
    const bf_t* __restrict__ Wc,                 // [4096][IN_DIM+HID] bf16
    const float* __restrict__ bias,              // [4096]
    float* __restrict__ c,                       // [64][1024] fp32, in-place
    bf_t* __restrict__ hout,                     // [64][1024] bf16 (ping-pong, swizzled)
    bf_t* __restrict__ hcopy, long copy_off)     // optional dec_out (swizzled)
{
    constexpr int K = IN_DIM + HID;
    const int lane = threadIdx.x & 31;
    const int half = lane >> 4;
    const int nl   = lane & 15;
    const int wm   = (blockIdx.x >> 5) * 2 + (threadIdx.x >> 5);  // m-tile 0..3
    const int jb   = (blockIdx.x & 31) * 32;                      // j-strip

    v8f acc[4][2];
#pragma unroll
    for (int g = 0; g < 4; ++g)
#pragma unroll
        for (int nt = 0; nt < 2; ++nt)
#pragma unroll
            for (int i = 0; i < 8; ++i) acc[g][nt][i] = 0.0f;

    const int r = wm * 16 + nl;                  // batch row this lane loads
    // hoisted per-lane row pointers (all inner loads become ptr+imm)
    const bf_t* wp[8];
#pragma unroll
    for (int g = 0; g < 4; ++g)
#pragma unroll
        for (int nt = 0; nt < 2; ++nt)
            wp[g * 2 + nt] = Wc + (long)(g * HID + jb + nt * 16 + nl) * K + half * 16;
    const bf_t* arow = Ax + (long)r * strideAx + half * 16;
    const bf_t* hrow = Ah + (long)r * HID      + half * 16;

    // ---- x phase ----
#pragma unroll 2
    for (int k0 = 0; k0 < IN_DIM; k0 += 32) {
        v16bf a = *reinterpret_cast<const v16bf*>(arow + k0);
#pragma unroll
        for (int i = 0; i < 8; ++i) {
            v16bf bf = *reinterpret_cast<const v16bf*>(wp[i] + k0);
            acc[i >> 1][i & 1] = __builtin_amdgcn_wmma_f32_16x16x32_bf16(
                false, a, false, bf, (short)0, acc[i >> 1][i & 1], false, false);
        }
    }
    // ---- h phase ----
#pragma unroll 2
    for (int k0 = 0; k0 < HID; k0 += 32) {
        v16bf a = *reinterpret_cast<const v16bf*>(hrow + k0);
#pragma unroll
        for (int i = 0; i < 8; ++i) {
            v16bf bf = *reinterpret_cast<const v16bf*>(wp[i] + IN_DIM + k0);
            acc[i >> 1][i & 1] = __builtin_amdgcn_wmma_f32_16x16x32_bf16(
                false, a, false, bf, (short)0, acc[i >> 1][i & 1], false, false);
        }
    }

    // ---- fused cell: i,f,g,o live in registers at matching slots ----
#pragma unroll
    for (int nt = 0; nt < 2; ++nt) {
        int j  = jb + nt * 16 + nl;
        int js = swz34(j);                        // swizzled store position
        float bi  = bias[j];
        float bfg = bias[j + 1024];
        float bg  = bias[j + 2048];
        float bo  = bias[j + 3072];
#pragma unroll
        for (int v = 0; v < 8; ++v) {
            int m = wm * 16 + half * 8 + v;       // C/D layout: VGPR v -> M = v + 8*half
            long cidx = (long)m * HID + j;
            float gi = acc[0][nt][v] + bi;
            float gf = acc[1][nt][v] + bfg;
            float gg = acc[2][nt][v] + bg;
            float go = acc[3][nt][v] + bo;
            float cn = sigm(gf) * c[cidx] + sigm(gi) * tanh_fast(gg);
            c[cidx] = cn;
            bf_t hb = f2bf(sigm(go) * tanh_fast(cn));
            hout[(long)m * HID + js] = hb;
            if (hcopy) hcopy[(long)m * (TDEC * HID) + copy_off + js] = hb;
        }
    }
}

// ---------------- FC projection GEMM (WMMA) ----------------
// preds[4032][512] = dec_out[4032][1024] @ fcW^T + fc_b ; wave = 1 m-tile x 2 n-tiles
__global__ __launch_bounds__(128) void fc_gemm(
    const bf_t* __restrict__ A,    // [4032][1024] bf16, K-swizzled
    const bf_t* __restrict__ W,    // [512][1024]  bf16 row-major
    const float* __restrict__ bias,// [512]
    float* __restrict__ out)       // [4032][512]
{
    const int lane = threadIdx.x & 31;
    const int half = lane >> 4, nl = lane & 15;
    const int wid  = blockIdx.x * 4 + (threadIdx.x >> 5);   // 0..4031
    const int mt   = wid >> 4;                              // 0..251
    const int nb   = (wid & 15) * 32;

    v8f acc[2];
#pragma unroll
    for (int nt = 0; nt < 2; ++nt)
#pragma unroll
        for (int i = 0; i < 8; ++i) acc[nt][i] = 0.0f;

    const bf_t* arow = A + (long)(mt * 16 + nl) * HID + half * 16;
    const bf_t* w0 = W + (long)(nb + nl) * HID + half * 16;
    const bf_t* w1 = W + (long)(nb + 16 + nl) * HID + half * 16;
#pragma unroll 2
    for (int k0 = 0; k0 < HID; k0 += 32) {
        v16bf a  = *reinterpret_cast<const v16bf*>(arow + k0);
        v16bf b0 = *reinterpret_cast<const v16bf*>(w0 + k0);
        v16bf b1 = *reinterpret_cast<const v16bf*>(w1 + k0);
        acc[0] = __builtin_amdgcn_wmma_f32_16x16x32_bf16(
            false, a, false, b0, (short)0, acc[0], false, false);
        acc[1] = __builtin_amdgcn_wmma_f32_16x16x32_bf16(
            false, a, false, b1, (short)0, acc[1], false, false);
    }
#pragma unroll
    for (int nt = 0; nt < 2; ++nt) {
        int n = nb + nt * 16 + nl;
        float bv = bias[n];
#pragma unroll
        for (int v = 0; v < 8; ++v) {
            int m = mt * 16 + half * 8 + v;
            out[(long)m * EMB + n] = acc[nt][v] + bv;
        }
    }
}

// ---------------- weight / data prep ----------------
__global__ void conv_weights(const float* __restrict__ Wih, const float* __restrict__ Whh,
                             const float* __restrict__ bih, const float* __restrict__ bhh,
                             int in_dim, int K, bf_t* __restrict__ Wc, float* __restrict__ bias)
{
    long idx = (long)blockIdx.x * blockDim.x + threadIdx.x;
    long total = (long)G4H * K;
    if (idx >= total) return;
    int n = (int)(idx / K), k = swz34((int)(idx % K));
    float v = (k < in_dim) ? Wih[(long)n * in_dim + k] : Whh[(long)n * HID + (k - in_dim)];
    Wc[idx] = f2bf(v);
    if (k == 0) bias[n] = bih[n] + bhh[n];
}

__global__ void conv_fcw(const float* __restrict__ W, bf_t* __restrict__ Wb) {
    long idx = (long)blockIdx.x * blockDim.x + threadIdx.x;
    if (idx < (long)EMB * HID) Wb[idx - (idx & 1023) + swz34((int)(idx & 1023))] = f2bf(W[idx]);
}

__global__ void gather_src(const int* __restrict__ src, const float* __restrict__ tab,
                           float* __restrict__ dstf, bf_t* __restrict__ dstb)
{
    long idx = (long)blockIdx.x * blockDim.x + threadIdx.x;   // 64*128*512
    if (idx >= (long)BN * SSRC * EMB) return;
    int e = (int)(idx & 511), t = (int)((idx >> 9) & 127), b = (int)(idx >> 16);
    int tok = src[b * SSRC + t];
    float v = tab[(long)tok * EMB + e];
    dstf[idx] = v;                                   // linear fp32 (attention)
    dstb[idx - e + swz34(e)] = f2bf(v);              // K-swizzled bf16 (GEMM A)
}

__global__ void gather_dec(const int* __restrict__ trg, const float* __restrict__ tab,
                           bf_t* __restrict__ dstb)
{
    long idx = (long)blockIdx.x * blockDim.x + threadIdx.x;   // 64*63*512
    if (idx >= (long)BN * TDEC * EMB) return;
    int e = (int)(idx % EMB);
    int t = (int)((idx / EMB) % TDEC);
    int b = (int)(idx / ((long)EMB * TDEC));
    int tok = (t == 0) ? 1 : trg[b * 64 + t];
    dstb[idx - e + swz34(e)] = f2bf(tab[(long)tok * EMB + e]);   // K-swizzled
}

__global__ void zero_words(unsigned int* __restrict__ p, long n) {
    long idx = (long)blockIdx.x * blockDim.x + threadIdx.x;
    if (idx < n) p[idx] = 0u;
}

// ---------------- attention ----------------
__global__ void scores_k(const float* __restrict__ preds, const float* __restrict__ srcf,
                         float* __restrict__ scores)
{
    int idx = blockIdx.x * blockDim.x + threadIdx.x;   // 64*63*128
    if (idx >= BN * TDEC * SSRC) return;
    int s = idx & 127, bt = idx >> 7;
    int b = bt / TDEC;
    const float4* p = (const float4*)(preds + (long)bt * EMB);
    const float4* q = (const float4*)(srcf + (long)b * SSRC * EMB + (long)s * EMB);
    float acc = 0.f;
    for (int e = 0; e < EMB / 4; ++e) {
        float4 a = p[e], c = q[e];
        acc += a.x * c.x + a.y * c.y + a.z * c.z + a.w * c.w;
    }
    scores[idx] = acc * 0.04419417382415922f;          // 1/sqrt(512)
}

__global__ __launch_bounds__(128) void softmax_k(float* __restrict__ sc,
                                                 float* __restrict__ out_attn)
{
    int lane = threadIdx.x & 31;
    int row  = blockIdx.x * 4 + (threadIdx.x >> 5);    // 0..4031
    float v[4];
#pragma unroll
    for (int i = 0; i < 4; ++i) v[i] = sc[(long)row * SSRC + lane + 32 * i];
    float m = fmaxf(fmaxf(v[0], v[1]), fmaxf(v[2], v[3]));
#pragma unroll
    for (int off = 16; off >= 1; off >>= 1) m = fmaxf(m, __shfl_xor(m, off, 32));
    float sum = 0.f;
#pragma unroll
    for (int i = 0; i < 4; ++i) { v[i] = __expf(v[i] - m); sum += v[i]; }
#pragma unroll
    for (int off = 16; off >= 1; off >>= 1) sum += __shfl_xor(sum, off, 32);
    float inv = 1.f / sum;
    int t = row % TDEC, b = row / TDEC;
#pragma unroll
    for (int i = 0; i < 4; ++i) {
        float a = v[i] * inv;
        sc[(long)row * SSRC + lane + 32 * i] = a;
        if (t == TDEC - 1) out_attn[b * SSRC + lane + 32 * i] = a;   // attn_w[:, -1, :]
    }
}

__global__ void ctx_k(const float* __restrict__ attn, const float* __restrict__ srcf,
                      float* __restrict__ out)
{
    int idx = blockIdx.x * blockDim.x + threadIdx.x;   // 64*64*512
    if (idx >= BN * 64 * EMB) return;
    int e = idx & 511, tp = (idx >> 9) & 63, b = idx >> 15;
    float r = 0.f;
    if (tp > 0) {
        const float* aw = attn + ((long)(b * TDEC + tp - 1)) * SSRC;
        const float* q  = srcf + (long)b * SSRC * EMB + e;
        for (int s = 0; s < SSRC; ++s) r += aw[s] * q[(long)s * EMB];
    }
    out[idx] = r;
}

// ---------------- host orchestration ----------------
extern "C" void kernel_launch(void* const* d_in, const int* in_sizes, int n_in,
                              void* d_out, int out_size, void* d_ws, size_t ws_size,
                              hipStream_t stream)
{
    const int*   src     = (const int*)  d_in[0];
    const int*   trg     = (const int*)  d_in[1];
    const float* src_tab = (const float*)d_in[2];
    const float* trg_tab = (const float*)d_in[3];
    const float* fcW     = (const float*)d_in[4];
    const float* fcb     = (const float*)d_in[5];
    // layer order: enc0, enc1, dec0, dec1
    const float* Wih[4] = {(const float*)d_in[6],  (const float*)d_in[10],
                           (const float*)d_in[14], (const float*)d_in[18]};
    const float* Whh[4] = {(const float*)d_in[7],  (const float*)d_in[11],
                           (const float*)d_in[15], (const float*)d_in[19]};
    const float* bih[4] = {(const float*)d_in[8],  (const float*)d_in[12],
                           (const float*)d_in[16], (const float*)d_in[20]};
    const float* bhh[4] = {(const float*)d_in[9],  (const float*)d_in[13],
                           (const float*)d_in[17], (const float*)d_in[21]};
    const int Kl[4]    = {EMB + HID, HID + HID, EMB + HID, HID + HID};
    const int indim[4] = {EMB, HID, EMB, HID};

    // ---- workspace carve-up (bytes) ----
    char* ws = (char*)d_ws;
    size_t off = 0;
    auto take = [&](size_t bytes) { char* p = ws + off; off += (bytes + 255) & ~(size_t)255; return p; };
    bf_t*  wc[4];
    wc[0] = (bf_t*)take((size_t)G4H * Kl[0] * 2);
    wc[1] = (bf_t*)take((size_t)G4H * Kl[1] * 2);
    wc[2] = (bf_t*)take((size_t)G4H * Kl[2] * 2);
    wc[3] = (bf_t*)take((size_t)G4H * Kl[3] * 2);
    float* bias   = (float*)take(4 * G4H * 4);
    bf_t*  fcwb   = (bf_t*) take((size_t)EMB * HID * 2);
    float* srcf   = (float*)take((size_t)BN * SSRC * EMB * 4);
    bf_t*  srcb   = (bf_t*) take((size_t)BN * SSRC * EMB * 2);
    bf_t*  decb   = (bf_t*) take((size_t)BN * TDEC * EMB * 2);
    bf_t*  hbuf   = (bf_t*) take((size_t)4 * BN * HID * 2);   // 2 layers x 2 parity, bf16
    float* cbuf   = (float*)take((size_t)2 * BN * HID * 4);   // c state, 2 layers, fp32
    bf_t*  decout = (bf_t*) take((size_t)BN * TDEC * HID * 2);
    float* preds  = (float*)take((size_t)BN * TDEC * EMB * 4);
    float* attn   = (float*)take((size_t)BN * TDEC * SSRC * 4);
    (void)ws_size; (void)in_sizes; (void)n_in; (void)out_size;

    bf_t* h0r = hbuf;                  bf_t* h0w = hbuf + 1 * BN * HID;
    bf_t* h1r = hbuf + 2 * BN * HID;   bf_t* h1w = hbuf + 3 * BN * HID;
    float* c0 = cbuf;                  float* c1 = cbuf + BN * HID;

    // ---- prep: weights -> bf16, embeddings, zero state ----
    for (int L = 0; L < 4; ++L) {
        long total = (long)G4H * Kl[L];
        conv_weights<<<(unsigned)((total + 255) / 256), 256, 0, stream>>>(
            Wih[L], Whh[L], bih[L], bhh[L], indim[L], Kl[L], wc[L], bias + L * G4H);
    }
    conv_fcw<<<(EMB * HID) / 256, 256, 0, stream>>>(fcW, fcwb);
    gather_src<<<(BN * SSRC * EMB) / 256, 256, 0, stream>>>(src, src_tab, srcf, srcb);
    gather_dec<<<(BN * TDEC * EMB + 255) / 256, 256, 0, stream>>>(trg, trg_tab, decb);
    zero_words<<<(4 * BN * HID * 2 / 4 + 255) / 256, 256, 0, stream>>>((unsigned*)hbuf, 4L * BN * HID * 2 / 4);
    zero_words<<<(2 * BN * HID + 255) / 256, 256, 0, stream>>>((unsigned*)cbuf, 2L * BN * HID);

    // ---- encoder: 128 steps, fused GEMM+cell, ping-pong h buffers ----
    for (int t = 0; t < SSRC; ++t) {
        lstm_step_fused<EMB><<<64, 64, 0, stream>>>(srcb + (long)t * EMB, (long)SSRC * EMB,
                                                    h0r, wc[0], bias + 0 * G4H,
                                                    c0, h0w, (bf_t*)nullptr, 0);
        lstm_step_fused<HID><<<64, 64, 0, stream>>>(h0w, (long)HID,
                                                    h1r, wc[1], bias + 1 * G4H,
                                                    c1, h1w, (bf_t*)nullptr, 0);
        bf_t* tmp;
        tmp = h0r; h0r = h0w; h0w = tmp;
        tmp = h1r; h1r = h1w; h1w = tmp;
    }
    // ---- decoder: 63 steps, h/c carry over (dec h0,c0 = enc hid,cell) ----
    for (int t = 0; t < TDEC; ++t) {
        lstm_step_fused<EMB><<<64, 64, 0, stream>>>(decb + (long)t * EMB, (long)TDEC * EMB,
                                                    h0r, wc[2], bias + 2 * G4H,
                                                    c0, h0w, (bf_t*)nullptr, 0);
        lstm_step_fused<HID><<<64, 64, 0, stream>>>(h0w, (long)HID,
                                                    h1r, wc[3], bias + 3 * G4H,
                                                    c1, h1w, decout, (long)t * HID);
        bf_t* tmp;
        tmp = h0r; h0r = h0w; h0w = tmp;
        tmp = h1r; h1r = h1w; h1w = tmp;
    }

    // ---- FC + attention ----
    fc_gemm<<<1008, 128, 0, stream>>>(decout, fcwb, fcb, preds);           // 4032 waves
    scores_k<<<(BN * TDEC * SSRC) / 256, 256, 0, stream>>>(preds, srcf, attn);
    softmax_k<<<1008, 128, 0, stream>>>(attn, (float*)d_out + (long)BN * 64 * EMB);
    ctx_k<<<(BN * 64 * EMB) / 256, 256, 0, stream>>>(attn, srcf, (float*)d_out);
}
